// FusionModule_84550726189740
// MI455X (gfx1250) — matrix operand order repaired and checked by reading, not compiled, not run on hardware
//
#include <hip/hip_runtime.h>

#define I_ITEMS 50000
#define DIM 512
#define EPS_SINK 0.01f
#define MAX_ITER 100
#define SCALE 100.0f
#define LN_EPS 1e-5f

typedef __bf16 bf16;
typedef __attribute__((ext_vector_type(16))) __bf16 v16bf;
typedef __attribute__((ext_vector_type(8)))  __bf16 v8bf;
typedef __attribute__((ext_vector_type(4)))  __bf16 v4bf;
typedef __attribute__((ext_vector_type(2)))  __bf16 v2bf;
typedef __attribute__((ext_vector_type(8)))  float   v8f;
typedef __attribute__((ext_vector_type(4)))  float   v4f;

// ---------------------------------------------------------------------------
// Kernel 1: per-column sum-of-squares accumulators
// ---------------------------------------------------------------------------
__global__ __launch_bounds__(256) void colsumsq_kernel(
    const float* __restrict__ zi, const float* __restrict__ zt,
    const float* __restrict__ zc,
    float* __restrict__ sq_img, float* __restrict__ sq_txt,
    float* __restrict__ sq_cf) {
  const int c  = threadIdx.x * 2;          // 2 columns per thread (512/256)
  const int r0 = blockIdx.x * 128;
  float si0 = 0.f, si1 = 0.f, st0 = 0.f, st1 = 0.f, sc0 = 0.f, sc1 = 0.f;
  for (int r = 0; r < 128; ++r) {
    const int i = r0 + r;
    if (i >= I_ITEMS) break;
    const size_t off = (size_t)i * DIM + c;
    float2 a = *(const float2*)(zi + off); si0 += a.x * a.x; si1 += a.y * a.y;
    float2 b = *(const float2*)(zt + off); st0 += b.x * b.x; st1 += b.y * b.y;
    float2 d = *(const float2*)(zc + off); sc0 += d.x * d.x; sc1 += d.y * d.y;
  }
  atomicAdd(&sq_img[c], si0); atomicAdd(&sq_img[c + 1], si1);
  atomicAdd(&sq_txt[c], st0); atomicAdd(&sq_txt[c + 1], st1);
  atomicAdd(&sq_cf[c],  sc0); atomicAdd(&sq_cf[c + 1],  sc1);
}

// ---------------------------------------------------------------------------
// Kernel 2: cross = A^T * B over items (split-K, WMMA bf16, f32 accumulate)
//   block = 256 threads (8 waves), tile 64(a) x 64(b), K-chunk 2048 items
//   Staging: each thread owns a k-row PAIR x 4 dims -> packed b32 LDS stores
// ---------------------------------------------------------------------------
#define KC 2048
#define CT 64
#define LDSK 40   // padded k-stride (80 bytes)

__global__ __launch_bounds__(256) void crossgemm_kernel(
    const float* __restrict__ A,   // [I, 512]
    const float* __restrict__ B,   // [I, 512]  (z_cf)
    float* __restrict__ acc) {     // [512, 512] f32 accumulator
  __shared__ __align__(16) bf16 As[CT][LDSK];  // [dim][k] transposed slabs
  __shared__ __align__(16) bf16 Bs[CT][LDSK];

  const int a0  = blockIdx.x * CT;
  const int b0  = blockIdx.y * CT;
  const int kg0 = blockIdx.z * KC;
  const int tid = threadIdx.x;
  const int lane = tid & 31;
  const int wave = tid >> 5;
  const int mt = wave & 3;                  // M tile 0..3
  const int ntBase = (wave >> 2) * 2;       // N tiles ntBase, ntBase+1
  const int lm = lane & 15, khalf = lane >> 4;

  // loader mapping: thread -> k-row pair kk (0..15), dim group d0 (0..60 step 4)
  const int kk = tid >> 4;                  // pair index
  const int d0 = (tid & 15) * 4;

  v8f acc0 = {}; v8f acc1 = {};

#define CROSS_STEP(GUARDED)                                                   \
  {                                                                           \
    const int it0 = kg0 + ks + 2 * kk;                                        \
    v4f aA, aB, bA, bB;                                                       \
    if (!(GUARDED) || (it0 < I_ITEMS)) {                                      \
      aA = *(const v4f*)(A + (size_t)it0 * DIM + a0 + d0);                    \
      bA = *(const v4f*)(B + (size_t)it0 * DIM + b0 + d0);                    \
    } else { aA = (v4f){}; bA = (v4f){}; }                                    \
    if (!(GUARDED) || (it0 + 1 < I_ITEMS)) {                                  \
      aB = *(const v4f*)(A + (size_t)(it0 + 1) * DIM + a0 + d0);              \
      bB = *(const v4f*)(B + (size_t)(it0 + 1) * DIM + b0 + d0);              \
    } else { aB = (v4f){}; bB = (v4f){}; }                                    \
    __syncthreads();                                                          \
    _Pragma("unroll")                                                         \
    for (int j = 0; j < 4; ++j) {                                             \
      v2bf pa = {(bf16)aA[j], (bf16)aB[j]};                                   \
      v2bf pb = {(bf16)bA[j], (bf16)bB[j]};                                   \
      *(v2bf*)&As[d0 + j][2 * kk] = pa;                                       \
      *(v2bf*)&Bs[d0 + j][2 * kk] = pb;                                       \
    }                                                                         \
    __syncthreads();                                                          \
    union { v16bf v; v8bf h[2]; } ua;                                         \
    {                                                                         \
      const bf16* pa = &As[mt * 16 + lm][0];                                  \
      ua.h[0] = *(const v8bf*)(pa + khalf * 16);                              \
      ua.h[1] = *(const v8bf*)(pa + khalf * 16 + 8);                          \
    }                                                                         \
    union { v16bf v; v8bf h[2]; } ub0, ub1;                                   \
    {                                                                         \
      const bf16* p0 = &Bs[ntBase * 16 + lm][khalf * 16];                     \
      ub0.h[0] = *(const v8bf*)p0; ub0.h[1] = *(const v8bf*)(p0 + 8);         \
      const bf16* p1 = &Bs[(ntBase + 1) * 16 + lm][khalf * 16];               \
      ub1.h[0] = *(const v8bf*)p1; ub1.h[1] = *(const v8bf*)(p1 + 8);         \
    }                                                                         \
    acc0 = __builtin_amdgcn_wmma_f32_16x16x32_bf16(false, ua.v, false, ub0.v, \
                                                   (short)0, acc0, false, false); \
    acc1 = __builtin_amdgcn_wmma_f32_16x16x32_bf16(false, ua.v, false, ub1.v, \
                                                   (short)0, acc1, false, false); \
  }

  if (kg0 + KC <= I_ITEMS) {
    for (int ks = 0; ks < KC; ks += 32) CROSS_STEP(false)
  } else {
    for (int ks = 0; ks < KC; ks += 32) CROSS_STEP(true)
  }
#undef CROSS_STEP

  // C/D layout: lane 0-15 -> n, VGPR r -> m = r + 8*(lane>>4)
#pragma unroll
  for (int r = 0; r < 8; ++r) {
    const int m = a0 + mt * 16 + r + 8 * khalf;
    const int n = b0 + ntBase * 16 + lm;
    atomicAdd(&acc[(size_t)m * 512 + n],      acc0[r]);
    atomicAdd(&acc[(size_t)m * 512 + n + 16], acc1[r]);
  }
}

// ---------------------------------------------------------------------------
// Kernel 3: Sinkhorn (one block per alignment), writes W^T in bf16 pre-scaled
// ---------------------------------------------------------------------------
__global__ __launch_bounds__(512) void sinkhorn_kernel(
    const float* __restrict__ crossImg, const float* __restrict__ crossTxt,
    const float* __restrict__ sq_img, const float* __restrict__ sq_txt,
    const float* __restrict__ sq_cf,
    const float* __restrict__ delta_img, const float* __restrict__ delta_txt,
    const float* __restrict__ alpha_p, const float* __restrict__ beta_p,
    float* __restrict__ Kmat_base, float* __restrict__ KmatT_base,
    bf16* __restrict__ Wt_img, bf16* __restrict__ Wt_txt) {
  const int al = blockIdx.x;
  const float* cross = (al == 0) ? crossImg : crossTxt;
  const float* sqmm  = (al == 0) ? sq_img : sq_txt;
  const float* delta = (al == 0) ? delta_img : delta_txt;
  const float wsc    = (al == 0) ? alpha_p[0] : beta_p[0];
  bf16* Wt   = (al == 0) ? Wt_img : Wt_txt;
  float* Km  = Kmat_base  + (size_t)al * 512 * 512;
  float* KmT = KmatT_base + (size_t)al * 512 * 512;

  __shared__ float u_s[512], v_s[512];
  const int t = threadIdx.x;
  const float invI = 1.0f / (float)I_ITEMS;
  const float coef = -SCALE / EPS_SINK;     // exponent = coef*(mm2+st2-2*cross)

  const float sa = sqmm[t] * invI;
  for (int b = 0; b < 512; ++b) {
    const float c = sa + sq_cf[b] * invI - 2.0f * cross[(size_t)t * 512 + b] * invI;
    const float k = __expf(coef * c);
    Km[(size_t)t * 512 + b]  = k;
    KmT[(size_t)b * 512 + t] = k;
  }
  const float mu = 1.0f / 512.0f, nu = 1.0f / 512.0f;
  u_s[t] = mu; v_s[t] = nu;
  __syncthreads();

  for (int it = 0; it < MAX_ITER; ++it) {
    float s = 0.f;
    const float* row = Km + (size_t)t * 512;
    for (int b = 0; b < 512; ++b) s += row[b] * v_s[b];
    u_s[t] = mu / (s + 1e-8f);
    __syncthreads();
    float s2 = 0.f;
    const float* col = KmT + (size_t)t * 512;
    for (int a = 0; a < 512; ++a) s2 += col[a] * u_s[a];
    v_s[t] = nu / (s2 + 1e-8f);
    __syncthreads();
  }

  // W[k][n] = u[k]*K[k][n]*v[n] + delta[k][n];  store column-major, pre-scaled
  const float uk = u_s[t];
  for (int n = 0; n < 512; ++n) {
    const float w = uk * Km[(size_t)t * 512 + n] * v_s[n] + delta[(size_t)t * 512 + n];
    Wt[(size_t)n * 512 + t] = (bf16)(wsc * w);
  }
}

// ---------------------------------------------------------------------------
// Kernel 4: out = normalize(LN((1-a-b)*z_cf + z_img@Wimg + z_txt@Wtxt))
//   (alpha/beta folded into Wt). 16 rows per block, 8 waves x 4 N-tiles
// ---------------------------------------------------------------------------
__global__ __launch_bounds__(256) void fuse_out_kernel(
    const float* __restrict__ zc, const float* __restrict__ zi,
    const float* __restrict__ zt,
    const bf16* __restrict__ WtImg, const bf16* __restrict__ WtTxt,
    const float* __restrict__ alpha_p, const float* __restrict__ beta_p,
    const float* __restrict__ ln_w, const float* __restrict__ ln_b,
    float* __restrict__ out) {
  __shared__ float zb[16][520];
  __shared__ float red[2][16][16];
  __shared__ float rowstat[16][2];

  const int i0  = blockIdx.x * 16;
  const int tid = threadIdx.x;
  const int wave = tid >> 5, lane = tid & 31;
  const int lm = lane & 15, khalf = lane >> 4;
  const int nBase = wave * 64;

  v8f acc[4] = {(v8f){}, (v8f){}, (v8f){}, (v8f){}};
  const float* rowImg = zi + (size_t)(i0 + lm) * DIM;
  const float* rowTxt = zt + (size_t)(i0 + lm) * DIM;

  for (int k0 = 0; k0 < DIM; k0 += 32) {
    // A fragments direct from f32 rows; packed f32->bf16 via convertvector
    union { v16bf v; v4bf q[4]; } uai, uat;
    uai.q[0] = __builtin_convertvector(*(const v4f*)(rowImg + k0 + khalf * 8), v4bf);
    uai.q[1] = __builtin_convertvector(*(const v4f*)(rowImg + k0 + khalf * 8 + 4), v4bf);
    uai.q[2] = __builtin_convertvector(*(const v4f*)(rowImg + k0 + 16 + khalf * 8), v4bf);
    uai.q[3] = __builtin_convertvector(*(const v4f*)(rowImg + k0 + 16 + khalf * 8 + 4), v4bf);
    uat.q[0] = __builtin_convertvector(*(const v4f*)(rowTxt + k0 + khalf * 8), v4bf);
    uat.q[1] = __builtin_convertvector(*(const v4f*)(rowTxt + k0 + khalf * 8 + 4), v4bf);
    uat.q[2] = __builtin_convertvector(*(const v4f*)(rowTxt + k0 + 16 + khalf * 8), v4bf);
    uat.q[3] = __builtin_convertvector(*(const v4f*)(rowTxt + k0 + 16 + khalf * 8 + 4), v4bf);

#pragma unroll
    for (int tt = 0; tt < 4; ++tt) {
      const int n = nBase + tt * 16 + lm;
      union { v16bf v; v8bf h[2]; } bi, bt;
      const bf16* pI = WtImg + (size_t)n * 512 + k0 + khalf * 8;
      bi.h[0] = *(const v8bf*)pI; bi.h[1] = *(const v8bf*)(pI + 16);
      const bf16* pT = WtTxt + (size_t)n * 512 + k0 + khalf * 8;
      bt.h[0] = *(const v8bf*)pT; bt.h[1] = *(const v8bf*)(pT + 16);
      acc[tt] = __builtin_amdgcn_wmma_f32_16x16x32_bf16(false, uai.v, false, bi.v,
                                                        (short)0, acc[tt], false, false);
      acc[tt] = __builtin_amdgcn_wmma_f32_16x16x32_bf16(false, uat.v, false, bt.v,
                                                        (short)0, acc[tt], false, false);
    }
  }

  // scatter accumulators to LDS tile
#pragma unroll
  for (int tt = 0; tt < 4; ++tt)
#pragma unroll
    for (int r = 0; r < 8; ++r)
      zb[r + 8 * khalf][nBase + tt * 16 + lm] = acc[tt][r];
  __syncthreads();

  // epilogue: blend + LayerNorm + L2 normalize (16 threads per row)
  const float a = alpha_p[0], b = beta_p[0];
  const float cfw = 1.0f - a - b;
  const int r   = tid >> 4;
  const int seg = (tid & 15) * 32;
  const float* cfrow = zc + (size_t)(i0 + r) * DIM;

  float sum = 0.f, sq = 0.f;
#pragma unroll 4
  for (int c = 0; c < 32; ++c) {
    const float v = zb[r][seg + c] + cfw * cfrow[seg + c];
    zb[r][seg + c] = v;
    sum += v; sq += v * v;
  }
  red[0][r][tid & 15] = sum; red[1][r][tid & 15] = sq;
  __syncthreads();
  if (tid < 16) {
    float s = 0.f, q = 0.f;
    for (int j = 0; j < 16; ++j) { s += red[0][tid][j]; q += red[1][tid][j]; }
    const float mean = s * (1.0f / 512.0f);
    const float var  = q * (1.0f / 512.0f) - mean * mean;
    rowstat[tid][0] = mean;
    rowstat[tid][1] = rsqrtf(var + LN_EPS);
  }
  __syncthreads();
  const float mean = rowstat[r][0], rstd = rowstat[r][1];
  float sq2 = 0.f;
#pragma unroll 4
  for (int c = 0; c < 32; ++c) {
    const float y = (zb[r][seg + c] - mean) * rstd * ln_w[seg + c] + ln_b[seg + c];
    zb[r][seg + c] = y;
    sq2 += y * y;
  }
  __syncthreads();                 // red[] reuse
  red[0][r][tid & 15] = sq2;
  __syncthreads();
  if (tid < 16) {
    float q = 0.f;
    for (int j = 0; j < 16; ++j) q += red[0][tid][j];
    rowstat[tid][0] = 1.0f / fmaxf(sqrtf(q), 1e-12f);
  }
  __syncthreads();
  const float rinv = rowstat[r][0];
  float* orow = out + (size_t)(i0 + r) * DIM;
#pragma unroll 4
  for (int c = 0; c < 32; ++c) orow[seg + c] = zb[r][seg + c] * rinv;
}

// ---------------------------------------------------------------------------
extern "C" void kernel_launch(void* const* d_in, const int* in_sizes, int n_in,
                              void* d_out, int out_size, void* d_ws, size_t ws_size,
                              hipStream_t stream) {
  const float* z_cf      = (const float*)d_in[0];
  const float* z_img     = (const float*)d_in[1];
  const float* z_txt     = (const float*)d_in[2];
  const float* delta_img = (const float*)d_in[3];
  const float* delta_txt = (const float*)d_in[4];
  const float* alpha     = (const float*)d_in[5];
  const float* beta      = (const float*)d_in[6];
  const float* ln_w      = (const float*)d_in[7];
  const float* ln_b      = (const float*)d_in[8];
  float* out = (float*)d_out;

  // workspace layout (~7 MB)
  float* sq_img   = (float*)d_ws;                    // 512
  float* sq_txt   = sq_img + 512;                    // 512
  float* sq_cf    = sq_txt + 512;                    // 512
  float* crossImg = sq_cf + 512;                     // 512*512
  float* crossTxt = crossImg + 512 * 512;            // 512*512
  float* Kmat     = crossTxt + 512 * 512;            // 2*512*512
  float* KmatT    = Kmat + 2 * 512 * 512;            // 2*512*512
  bf16*  WtImg    = (bf16*)(KmatT + 2 * 512 * 512);  // 512*512 bf16
  bf16*  WtTxt    = WtImg + 512 * 512;               // 512*512 bf16

  // zero atomic accumulators (graph-capturable)
  hipMemsetAsync(d_ws, 0, (size_t)(3 * 512 + 2 * 512 * 512) * sizeof(float), stream);

  colsumsq_kernel<<<dim3((I_ITEMS + 127) / 128), 256, 0, stream>>>(
      z_img, z_txt, z_cf, sq_img, sq_txt, sq_cf);

  dim3 cg(512 / CT, 512 / CT, (I_ITEMS + KC - 1) / KC);
  crossgemm_kernel<<<cg, 256, 0, stream>>>(z_img, z_cf, crossImg);
  crossgemm_kernel<<<cg, 256, 0, stream>>>(z_txt, z_cf, crossTxt);

  sinkhorn_kernel<<<2, 512, 0, stream>>>(crossImg, crossTxt, sq_img, sq_txt, sq_cf,
                                         delta_img, delta_txt, alpha, beta,
                                         Kmat, KmatT, WtImg, WtTxt);

  fuse_out_kernel<<<I_ITEMS / 16, 256, 0, stream>>>(
      z_cf, z_img, z_txt, WtImg, WtTxt, alpha, beta, ln_w, ln_b, out);
}
